// GroupedQueryAttention_41343355191299
// MI455X (gfx1250) — matrix operand order repaired and checked by reading, not compiled, not run on hardware
//
#include <hip/hip_runtime.h>
#include <hip/hip_bf16.h>

// ---------------------------------------------------------------------------
// GQA attention block for MI455X (gfx1250), all contractions via
// v_wmma_f32_16x16x32_bf16 (wave32, f32 accumulate).
// ---------------------------------------------------------------------------

typedef __bf16 bf16_t;
typedef __attribute__((ext_vector_type(16))) __bf16 v16bf;
typedef __attribute__((ext_vector_type(8)))  __bf16 v8bf;
typedef __attribute__((ext_vector_type(8)))  float  v8f;

#define NHEAD   24
#define NKV     6
#define HD      40
#define DPAD    64
#define BB      2
#define TT      2048
#define CC      960
#define QKV_OUT 1440
#define BT      (BB * TT)      // 4096

#define WMMA_BF16(a, b, c) \
  __builtin_amdgcn_wmma_f32_16x16x32_bf16(false, (a), false, (b), (short)0, (c), false, false)

static __device__ __forceinline__ v16bf cat8(v8bf lo, v8bf hi) {
  return __builtin_shufflevector(lo, hi, 0,1,2,3,4,5,6,7,8,9,10,11,12,13,14,15);
}

// A-fragment (16x32 bf16, M x K). rowptr already includes the lane's row and
// the lane-half K offset; elements 0..7 come from +0, 8..15 from +8.
static __device__ __forceinline__ v16bf load_a_frag(const bf16_t* rowptr) {
  v8bf lo = *(const v8bf*)(rowptr);
  v8bf hi = *(const v8bf*)(rowptr + 8);
  return cat8(lo, hi);
}

// ---------------------------------------------------------------------------
// f32 -> bf16 conversion
// ---------------------------------------------------------------------------
__global__ void f32_to_bf16_k(const float* __restrict__ in, bf16_t* __restrict__ out, int n) {
  int i = blockIdx.x * blockDim.x + threadIdx.x;
  if (i < n) out[i] = (bf16_t)in[i];
}

// ---------------------------------------------------------------------------
// GEMM: C[m,n] = sum_k A[m,k] * W[n,k] + bias[n]
// A: [M,K] bf16 row-major, W: [N,K] bf16 row-major, C: [M,N] f32.
// One wave -> 32(M) x 48(N) tile: 2 A-frags x 3 B-frags = 6 WMMAs per K-step
// of 32, with each B fragment reused across both row tiles (~20 FLOP/B from
// L2, vs ~12 for a 16x48 tile). Operands are L2-resident (x_bf16 7.9MB +
// weights 4.6MB << 192MB L2), so fragments are loaded straight from global
// with A and B both in the same contiguous k order (k = 16*half + s).
// ---------------------------------------------------------------------------
__global__ __launch_bounds__(256)
void gemm_bf16_f32(const bf16_t* __restrict__ A, const bf16_t* __restrict__ W,
                   const float* __restrict__ bias, float* __restrict__ C,
                   int M, int N, int K) {
  const int lane = threadIdx.x & 31;
  const int wave = threadIdx.x >> 5;
  const int mt   = blockIdx.y * 8 + wave;   // 32-row tile index
  const int m0   = mt * 32;
  if (m0 >= M) return;
  const int n0   = blockIdx.x * 48;
  const int row  = lane & 15;
  const bool hi  = lane >= 16;

  const bf16_t* arow0 = A + (size_t)(m0 + row) * K + (hi ? 16 : 0);
  const bf16_t* arow1 = A + (size_t)(m0 + 16 + row) * K + (hi ? 16 : 0);
  const bf16_t* wrow  = W + (size_t)(n0 + row) * K + (hi ? 16 : 0);

  v8f acc00 = {}, acc01 = {}, acc02 = {};
  v8f acc10 = {}, acc11 = {}, acc12 = {};
  for (int k = 0; k < K; k += 32) {
    v16bf a0 = load_a_frag(arow0 + k);
    v16bf a1 = load_a_frag(arow1 + k);
    v16bf b0 = *(const v16bf*)(wrow + k);
    v16bf b1 = *(const v16bf*)(wrow + (size_t)16 * K + k);
    v16bf b2 = *(const v16bf*)(wrow + (size_t)32 * K + k);
    acc00 = WMMA_BF16(a0, b0, acc00);
    acc01 = WMMA_BF16(a0, b1, acc01);
    acc02 = WMMA_BF16(a0, b2, acc02);
    acc10 = WMMA_BF16(a1, b0, acc10);
    acc11 = WMMA_BF16(a1, b1, acc11);
    acc12 = WMMA_BF16(a1, b2, acc12);
  }

  const int nlo = lane & 15;
  const float bv0 = bias[n0 + nlo];
  const float bv1 = bias[n0 + 16 + nlo];
  const float bv2 = bias[n0 + 32 + nlo];
#pragma unroll
  for (int r = 0; r < 8; ++r) {
    const int mr = r + (hi ? 8 : 0);
    float* crow0 = C + (size_t)(m0 + mr) * N + n0;
    crow0[nlo]      = acc00[r] + bv0;
    crow0[16 + nlo] = acc01[r] + bv1;
    crow0[32 + nlo] = acc02[r] + bv2;
    float* crow1 = C + (size_t)(m0 + 16 + mr) * N + n0;
    crow1[nlo]      = acc10[r] + bv0;
    crow1[16 + nlo] = acc11[r] + bv1;
    crow1[32 + nlo] = acc12[r] + bv2;
  }
}

// ---------------------------------------------------------------------------
// RoPE (traditional, interleaved pairs) + scatter qkv[B,T,1440] (f32, bias
// already added by the GEMM) into:
//   Qb [B, NHEAD, T, DPAD] bf16    (d >= HD pre-zeroed)
//   Kb [B, NKV,  T, DPAD] bf16     (d >= HD pre-zeroed)
//   Vt [B, NKV, DPAD, T]  bf16     (transposed so attn.V B-frags are contiguous)
// One thread per output pair (720 pairs per token).
// ---------------------------------------------------------------------------
__global__ void rope_scatter_k(const float* __restrict__ qkv,
                               bf16_t* __restrict__ Qb, bf16_t* __restrict__ Kb,
                               bf16_t* __restrict__ Vt) {
  int idx = blockIdx.x * blockDim.x + threadIdx.x;
  if (idx >= BB * TT * (QKV_OUT / 2)) return;
  int d2 = idx % (QKV_OUT / 2);
  int bt = idx / (QKV_OUT / 2);
  int t  = bt % TT;
  int b  = bt / TT;
  const float* src = qkv + (size_t)bt * QKV_OUT;
  int o = d2 * 2;

  if (o < NHEAD * HD) {                       // ---- Q (RoPE) ----
    int h = o / HD, dd = o % HD;
    float x1 = src[o], x2 = src[o + 1];
    float inv = __expf(-(float)dd * (9.210340371976184f / (float)HD)); // base^(-dd/HD), dd even
    float s, c;  __sincosf((float)t * inv, &s, &c);
    bf16_t* dst = Qb + ((((size_t)b * NHEAD + h) * TT + t) * DPAD) + dd;
    dst[0] = (bf16_t)(x1 * c - x2 * s);
    dst[1] = (bf16_t)(x1 * s + x2 * c);
  } else if (o < NHEAD * HD + NKV * HD) {     // ---- K (RoPE) ----
    int oo = o - NHEAD * HD;
    int h = oo / HD, dd = oo % HD;
    float x1 = src[o], x2 = src[o + 1];
    float inv = __expf(-(float)dd * (9.210340371976184f / (float)HD));
    float s, c;  __sincosf((float)t * inv, &s, &c);
    bf16_t* dst = Kb + ((((size_t)b * NKV + h) * TT + t) * DPAD) + dd;
    dst[0] = (bf16_t)(x1 * c - x2 * s);
    dst[1] = (bf16_t)(x1 * s + x2 * c);
  } else {                                    // ---- V (transpose store) ----
    int oo = o - NHEAD * HD - NKV * HD;
    int h = oo / HD, dd = oo % HD;
    bf16_t* base = Vt + (((size_t)b * NKV + h) * DPAD) * TT + t;
    base[(size_t)dd * TT]       = (bf16_t)src[o];
    base[(size_t)(dd + 1) * TT] = (bf16_t)src[o + 1];
  }
}

// ---------------------------------------------------------------------------
// Causal flash attention, one wave per 16-row Q tile of one (b,h).
// scores = Q(16xDPAD) . K^T  via 2 WMMAs per 16-col S tile (K split 0-31/32-63)
// online softmax on the f32 C-layout (row lives in a 16-lane half -> shfl_xor)
// P re-shaped C-layout -> A-layout via per-wave LDS (in-order within a wave)
// O += P(16x32) . V(32x48) via 3 WMMAs (d<48 covers head_dim 40)
// ---------------------------------------------------------------------------
__global__ __launch_bounds__(128)
void attn_k(const bf16_t* __restrict__ Qb, const bf16_t* __restrict__ Kb,
            const bf16_t* __restrict__ Vt, bf16_t* __restrict__ attn_out) {
  __shared__ bf16_t Pl[4][16 * 32];

  const int lane = threadIdx.x & 31;
  const int wv   = threadIdx.x >> 5;
  const int b    = blockIdx.z;
  const int h    = blockIdx.y;
  const int t0   = (blockIdx.x * 4 + wv) * 16;
  const int hk   = h / (NHEAD / NKV);
  const int row  = lane & 15;
  const bool hi  = lane >= 16;

  const bf16_t* Q  = Qb + (((size_t)b * NHEAD + h) * TT + t0) * DPAD;
  const bf16_t* Kh = Kb + (((size_t)b * NKV + hk) * TT) * DPAD;
  const bf16_t* Vh = Vt + (((size_t)b * NKV + hk) * DPAD) * TT;  // [DPAD][TT]

  // Q A-fragments for K-chunks d=0..31 and d=32..63
  const v16bf aq0 = load_a_frag(Q + (size_t)row * DPAD + (hi ? 16 : 0));
  const v16bf aq1 = load_a_frag(Q + (size_t)row * DPAD + 32 + (hi ? 16 : 0));

  float Mrow[8], Lrow[8];
  v8f O[3] = {};
#pragma unroll
  for (int r = 0; r < 8; ++r) { Mrow[r] = -3.0e38f; Lrow[r] = 0.0f; }

  const float scale = 0.15811388300841897f;  // 1/sqrt(40)

  for (int s0 = 0; s0 < t0 + 16; s0 += 32) {
    // ---- scores: two 16x16 tiles ----
    v8f sc0 = {}, sc1 = {};
    {
      const bf16_t* kp0 = Kh + (size_t)(s0 + row) * DPAD + (hi ? 16 : 0);
      const bf16_t* kp1 = Kh + (size_t)(s0 + 16 + row) * DPAD + (hi ? 16 : 0);
      v16bf b00 = *(const v16bf*)(kp0);
      v16bf b01 = *(const v16bf*)(kp0 + 32);
      v16bf b10 = *(const v16bf*)(kp1);
      v16bf b11 = *(const v16bf*)(kp1 + 32);
      sc0 = WMMA_BF16(aq0, b00, sc0);
      sc0 = WMMA_BF16(aq1, b01, sc0);
      sc1 = WMMA_BF16(aq0, b10, sc1);
      sc1 = WMMA_BF16(aq1, b11, sc1);
    }

    // ---- online softmax (row = 16-lane half of the C-layout) ----
    const int s_a = s0 + row;
#pragma unroll
    for (int r = 0; r < 8; ++r) {
      const int t_g = t0 + r + (hi ? 8 : 0);
      float v0 = sc0[r] * scale;  v0 = (s_a      > t_g) ? -3.0e38f : v0;
      float v1 = sc1[r] * scale;  v1 = (s_a + 16 > t_g) ? -3.0e38f : v1;
      float m = fmaxf(v0, v1);
#pragma unroll
      for (int off = 8; off; off >>= 1) m = fmaxf(m, __shfl_xor(m, off, 32));
      float mn = fmaxf(Mrow[r], m);
      float al = __expf(Mrow[r] - mn);
      Mrow[r] = mn;
      float p0 = __expf(v0 - mn);
      float p1 = __expf(v1 - mn);
      float rs = p0 + p1;
#pragma unroll
      for (int off = 8; off; off >>= 1) rs += __shfl_xor(rs, off, 32);
      Lrow[r] = Lrow[r] * al + rs;
      O[0][r] *= al;  O[1][r] *= al;  O[2][r] *= al;
      const int mrow = r + (hi ? 8 : 0);
      Pl[wv][mrow * 32 + row]      = (bf16_t)p0;
      Pl[wv][mrow * 32 + 16 + row] = (bf16_t)p1;
    }

    // LDS is in-order within a wave (DScnt); only block compiler reordering.
    asm volatile("" ::: "memory");

    // ---- P (A-layout from LDS) . V ----
    const v16bf ap = load_a_frag(&Pl[wv][row * 32 + (hi ? 16 : 0)]);
#pragma unroll
    for (int j = 0; j < 3; ++j) {
      const bf16_t* vp = Vh + (size_t)(j * 16 + row) * TT + s0 + (hi ? 16 : 0);
      v16bf bv = *(const v16bf*)vp;
      O[j] = WMMA_BF16(ap, bv, O[j]);
    }
  }

  // ---- normalize + store bf16 rows for the projection GEMM ----
#pragma unroll
  for (int j = 0; j < 3; ++j) {
#pragma unroll
    for (int r = 0; r < 8; ++r) {
      const int d = j * 16 + row;
      if (d < HD) {
        const int t_g = t0 + r + (hi ? 8 : 0);
        float o = O[j][r] / Lrow[r];
        attn_out[((size_t)b * TT + t_g) * CC + h * HD + d] = (bf16_t)o;
      }
    }
  }
}

// ---------------------------------------------------------------------------
// Host-side orchestration
// ---------------------------------------------------------------------------
extern "C" void kernel_launch(void* const* d_in, const int* in_sizes, int n_in,
                              void* d_out, int out_size, void* d_ws, size_t ws_size,
                              hipStream_t stream) {
  (void)in_sizes; (void)n_in; (void)out_size; (void)ws_size;
  const float* x     = (const float*)d_in[0];
  const float* Wqkv  = (const float*)d_in[1];
  const float* bqkv  = (const float*)d_in[2];
  const float* Wproj = (const float*)d_in[3];
  const float* bproj = (const float*)d_in[4];
  float* out = (float*)d_out;

  char* ws = (char*)d_ws;
  size_t off = 0;
  auto take = [&](size_t bytes) -> void* {
    void* p = ws + off;
    off = (off + bytes + 255) & ~(size_t)255;
    return p;
  };

  const size_t n_x   = (size_t)BT * CC;
  const size_t n_wq  = (size_t)QKV_OUT * CC;
  const size_t n_wp  = (size_t)CC * CC;
  const size_t n_qkv = (size_t)BT * QKV_OUT;
  const size_t n_qb  = (size_t)BB * NHEAD * TT * DPAD;
  const size_t n_kb  = (size_t)BB * NKV * TT * DPAD;

  bf16_t* xb   = (bf16_t*)take(n_x * 2);
  bf16_t* wqb  = (bf16_t*)take(n_wq * 2);
  bf16_t* wpb  = (bf16_t*)take(n_wp * 2);
  float*  qkv  = (float*)take(n_qkv * 4);
  bf16_t* Qb   = (bf16_t*)take(n_qb * 2);
  bf16_t* Kb   = (bf16_t*)take(n_kb * 2);
  bf16_t* Vt   = (bf16_t*)take(n_kb * 2);
  bf16_t* aob  = (bf16_t*)take(n_x * 2);

  // 1) convert inputs to bf16 (one-time traffic; everything stays L2-resident)
  f32_to_bf16_k<<<(int)((n_x  + 255) / 256), 256, 0, stream>>>(x,     xb,  (int)n_x);
  f32_to_bf16_k<<<(int)((n_wq + 255) / 256), 256, 0, stream>>>(Wqkv,  wqb, (int)n_wq);
  f32_to_bf16_k<<<(int)((n_wp + 255) / 256), 256, 0, stream>>>(Wproj, wpb, (int)n_wp);

  // 2) QKV GEMM: [4096,960] x [1440,960]^T -> [4096,1440] f32 (+bias)
  gemm_bf16_f32<<<dim3(QKV_OUT / 48, BT / 256), 256, 0, stream>>>(
      xb, wqb, bqkv, qkv, BT, QKV_OUT, CC);

  // 3) zero the padded Q/K/V buffers, then RoPE + scatter
  hipMemsetAsync(Qb, 0, n_qb * 2, stream);
  hipMemsetAsync(Kb, 0, n_kb * 2, stream);
  hipMemsetAsync(Vt, 0, n_kb * 2, stream);
  {
    const int total = BB * TT * (QKV_OUT / 2);
    rope_scatter_k<<<(total + 255) / 256, 256, 0, stream>>>(qkv, Qb, Kb, Vt);
  }

  // 4) causal flash attention (wave per 16-row Q tile)
  attn_k<<<dim3(TT / 16 / 4, NHEAD, BB), 128, 0, stream>>>(Qb, Kb, Vt, aob);

  // 5) output projection: [4096,960] x [960,960]^T -> [4096,960] f32 (+bias)
  gemm_bf16_f32<<<dim3(CC / 48, BT / 256), 256, 0, stream>>>(
      aob, wpb, bproj, out, BT, CC, CC);
}
